// AtomEmbedding_84361747628495
// MI455X (gfx1250) — matrix operand order repaired and checked by reading, not compiled, not run on hardware
//
#include <hip/hip_runtime.h>

typedef _Float16 v16h __attribute__((ext_vector_type(16)));
typedef _Float16 v8h  __attribute__((ext_vector_type(8)));
typedef float    v8f  __attribute__((ext_vector_type(8)));

#define DM 256
#define KCH 8          // 256 / 32 K-chunks
#define TILES 16       // 256 / 16 column tiles
#define LDK 264        // padded w2^T row length in f16 (528B -> 4n mod 64 banks)
#define GROUPS 4       // 16-row groups per wave
#define ROWS_PER_BLOCK (8 * 16 * GROUPS)   // 512

// LDS layout (dynamic): w2t[256][LDK] f16 | w1s[3*256] | b1s[256] | b2s[256] | gs[256] | bs[256]
#define SMEM_BYTES (DM * LDK * 2 + (3 * DM + 4 * DM) * 4)

#define LOG2E 1.44269504088896340736f

extern "C" __global__ __launch_bounds__(256)
void atom_embed_mlp_ln(const int* __restrict__ z,
                       const float* __restrict__ x,
                       const float* __restrict__ emb,
                       const float* __restrict__ w1,
                       const float* __restrict__ b1,
                       const float* __restrict__ w2,
                       const float* __restrict__ b2,
                       const float* __restrict__ gamma,
                       const float* __restrict__ beta,
                       float* __restrict__ out,
                       int N)
{
    extern __shared__ char smem[];
    _Float16* w2t = (_Float16*)smem;
    float* w1s = (float*)(smem + DM * LDK * 2);
    float* b1s = w1s + 3 * DM;
    float* b2s = b1s + DM;
    float* gs  = b2s + DM;
    float* bs  = gs  + DM;

    const int tid = threadIdx.x;

    // ---- Phase 0: stage w2^T as f16 (coalesced global reads, 2-way max LDS bank conflict) ----
    {
        const int n = tid;
        for (int k = 0; k < DM; ++k)
            w2t[n * LDK + k] = (_Float16)w2[k * DM + n];
    }
    for (int i = tid; i < 3 * DM; i += 256) w1s[i] = w1[i];
    b1s[tid] = b1[tid];
    b2s[tid] = b2[tid];
    gs[tid]  = gamma[tid];
    bs[tid]  = beta[tid];
    __syncthreads();

    const int wave = tid >> 5;
    const int lane = tid & 31;
    const int ln   = lane & 15;
    const int half = lane >> 4;     // which 16-lane half of the wave

    for (int g = 0; g < GROUPS; ++g) {
        const int rowbase = blockIdx.x * ROWS_PER_BLOCK + (wave * GROUPS + g) * 16;
        if (rowbase + 16 > N) continue;

        // A-matrix row for this lane (A layout: lanes 0-15 and 16-31 both cover M=0..15)
        const int arow = rowbase + ln;
        const float x0 = x[arow * 3 + 0];
        const float x1 = x[arow * 3 + 1];
        const float x2 = x[arow * 3 + 2];

        v8f acc[TILES] = {};   // 16 tiles x 8 f32 accumulators

        #pragma unroll
        for (int c = 0; c < KCH; ++c) {
            // Build A fragment with VALU: silu(x @ w1 + b1), per ISA 16-bit A 16x32 layout.
            // e<8  -> K = 32c + 8*half + e        (VGPR0-3)
            // e>=8 -> K = 32c + 16 + 8*half + e-8 (VGPR4-7)
            v16h a;
            #pragma unroll
            for (int e = 0; e < 16; ++e) {
                const int k = 32 * c + ((e < 8) ? 0 : 16) + 8 * half + (e & 7);
                float v = fmaf(x0, w1s[k],
                          fmaf(x1, w1s[DM + k],
                          fmaf(x2, w1s[2 * DM + k], b1s[k])));
                // SiLU via raw HW transcendentals: v_exp_f32 + v_rcp_f32 (TRANS class,
                // co-executes with XDL WMMA)
                const float sig = __builtin_amdgcn_rcpf(
                    1.0f + __builtin_amdgcn_exp2f(v * (-LOG2E)));
                v = v * sig;
                a[e] = (_Float16)v;
            }
            #pragma unroll
            for (int t = 0; t < TILES; ++t) {
                // B fragment: column n = 16t + ln, K = 32c + 8*half + e (e<8), 32c + 16 + 8*half + e-8 (e>=8) from LDS (conflict-free pad)
                const _Float16* bp = &w2t[(t * 16 + ln) * LDK + 32 * c + 8 * half];
                union { v16h v; v8h h[2]; } bb;
                bb.h[0] = *(const v8h*)(bp);
                bb.h[1] = *(const v8h*)(bp + 16);
                acc[t] = __builtin_amdgcn_wmma_f32_16x16x32_f16(
                    false, a, false, bb.v, (short)0, acc[t], false, false);
            }
        }

        // ---- Fused epilogue: h = p2 + b2 + emb[z]; LayerNorm; store ----
        // C layout: VGPR j -> row j (lanes 0-15) / j+8 (lanes 16-31), N = lane%16
        int zr[8];
        #pragma unroll
        for (int j = 0; j < 8; ++j) zr[j] = z[rowbase + half * 8 + j];

        #pragma unroll
        for (int t = 0; t < TILES; ++t) {
            const int col = t * 16 + ln;
            const float b2v = b2s[col];
            #pragma unroll
            for (int j = 0; j < 8; ++j)
                acc[t][j] += b2v + emb[zr[j] * DM + col];
        }

        float mu[8], rs[8];
        #pragma unroll
        for (int j = 0; j < 8; ++j) {
            float s = 0.f, q = 0.f;
            #pragma unroll
            for (int t = 0; t < TILES; ++t) { const float h = acc[t][j]; s += h; q += h * h; }
            // reduce across the 16 lanes of this half (xor masks < 16 stay in-half)
            s += __shfl_xor(s, 1, 32);  q += __shfl_xor(q, 1, 32);
            s += __shfl_xor(s, 2, 32);  q += __shfl_xor(q, 2, 32);
            s += __shfl_xor(s, 4, 32);  q += __shfl_xor(q, 4, 32);
            s += __shfl_xor(s, 8, 32);  q += __shfl_xor(q, 8, 32);
            const float m = s * (1.0f / 256.0f);
            mu[j] = m;
            // raw v_rsq_f32
            rs[j] = __builtin_amdgcn_rsqf(q * (1.0f / 256.0f) - m * m + 1e-5f);
        }

        #pragma unroll
        for (int t = 0; t < TILES; ++t) {
            const int col = t * 16 + ln;
            const float gv = gs[col];
            const float bv = bs[col];
            #pragma unroll
            for (int j = 0; j < 8; ++j) {
                const int row = rowbase + half * 8 + j;
                // Write-once 512MB output: non-temporal store keeps w2/emb/x hot in L2
                __builtin_nontemporal_store(
                    (acc[t][j] - mu[j]) * rs[j] * gv + bv,
                    &out[row * DM + col]);
            }
        }
    }
}

extern "C" void kernel_launch(void* const* d_in, const int* in_sizes, int n_in,
                              void* d_out, int out_size, void* d_ws, size_t ws_size,
                              hipStream_t stream) {
    const int*   zp    = (const int*)  d_in[0];
    const float* xp    = (const float*)d_in[1];
    const float* embp  = (const float*)d_in[2];
    const float* w1p   = (const float*)d_in[3];
    const float* b1p   = (const float*)d_in[4];
    const float* w2p   = (const float*)d_in[5];
    const float* b2p   = (const float*)d_in[6];
    const float* gp    = (const float*)d_in[7];
    const float* bp    = (const float*)d_in[8];
    float* outp = (float*)d_out;

    const int N = in_sizes[0];
    const int blocks = (N + ROWS_PER_BLOCK - 1) / ROWS_PER_BLOCK;

    // >64KB dynamic LDS: opt in every call (deterministic, no stream op)
    (void)hipFuncSetAttribute(reinterpret_cast<const void*>(atom_embed_mlp_ln),
                              hipFuncAttributeMaxDynamicSharedMemorySize,
                              (int)SMEM_BYTES);

    atom_embed_mlp_ln<<<blocks, 256, SMEM_BYTES, stream>>>(
        zp, xp, embp, w1p, b1p, w2p, b2p, gp, bp, outp, N);
}
